// MoVELargeLayer_63513976373283
// MI455X (gfx1250) — matrix operand order repaired and checked by reading, not compiled, not run on hardware
//
#include <hip/hip_runtime.h>

typedef __attribute__((ext_vector_type(16))) __bf16 v16bf;
typedef __attribute__((ext_vector_type(8)))  float  v8f;

#define CF_B_BF16  1
#define CF_GELU    2
#define CF_ELU1    4
#define CF_ROWMUL  8
#define CF_ADDRES  16
#define CF_STOREF  32
#define CF_STOREBF 64
#define CF_GATHER  128
#define CF_SCATTER 256
#define CF_EXPERT  512

__device__ __forceinline__ float bf2f(unsigned short s) {
    unsigned u = ((unsigned)s) << 16;
    return __builtin_bit_cast(float, u);
}
__device__ __forceinline__ unsigned short f2bf(float f) {
    unsigned u = __builtin_bit_cast(unsigned, f);
    unsigned r = (u + 0x7fffu + ((u >> 16) & 1u)) >> 16;
    return (unsigned short)r;
}
__device__ __forceinline__ void put2(v16bf& v, int i, unsigned u) {
    v[i]     = __builtin_bit_cast(__bf16, (unsigned short)(u & 0xffffu));
    v[i + 1] = __builtin_bit_cast(__bf16, (unsigned short)(u >> 16));
}

// ---------------------------------------------------------------------------
// Generic bf16 WMMA GEMM: C = act(A[bf16] @ B[f32|bf16] + bias) (+rowmul/res).
// Macro-tile 64 x (32*JW), 256 threads (8 waves), K-step 32, double-buffered
// LDS. A-tiles move via GLOBAL_LOAD_ASYNC_TO_LDS_B128 (ASYNCcnt); B-tiles go
// through VGPRs for the f32->bf16 convert.
// ---------------------------------------------------------------------------
template<int CF, int JW>
__launch_bounds__(256)
__global__ void gemm_k(const unsigned short* __restrict__ A, long lda, long sAb, long sAh,
                       const void* __restrict__ Bsrc, long ldb, long sBb, long sBh,
                       const float* __restrict__ bias,
                       float* __restrict__ Cf, long ldc, long sCb, long sCh,
                       unsigned short* __restrict__ Cbf, long ldcb, long sCbb, long sCbh,
                       const float* __restrict__ rowptr, long rstride, long rb, long rh,
                       const float* __restrict__ res, long ldres,
                       int Kdim, int Hdiv,
                       const int* __restrict__ offv, const int* __restrict__ asg_tok,
                       const float* __restrict__ asg_w, const int* __restrict__ asg_slot,
                       float* __restrict__ ybuf, long yslot_stride,
                       long expB_stride, long expBias_stride)
{
    constexpr int NM = 32 * JW;                 // macro-tile width in N
    __shared__ __align__(16) unsigned short As[2][64][40];
    __shared__ __align__(16) unsigned short Bs[2][NM][40];
    const int tid = threadIdx.x, wv = tid >> 5, lane = tid & 31;
    const int L16 = lane & 15, hiL = lane >> 4;
    const long blockRow = (long)blockIdx.x * 64, blockCol = (long)blockIdx.y * NM;
    const int z = blockIdx.z, zb = z / Hdiv, zh = z % Hdiv;

    int eidx = 0;
    if constexpr (CF & CF_EXPERT) {
        if (blockRow >= (long)offv[8]) return;
        for (int e = 0; e < 8; ++e)
            if (blockRow >= (long)offv[e] && blockRow < (long)offv[e + 1]) eidx = e;
    }
    const unsigned short* Ab = A + (long)zb * sAb + (long)zh * sAh;
    const float* biasp = bias;
    long boff = (long)zb * sBb + (long)zh * sBh;
    if constexpr (CF & CF_EXPERT) {
        boff += (long)eidx * expB_stride;
        if (biasp) biasp += (long)eidx * expBias_stride;
    }
    const float*          Bf = (const float*)Bsrc + boff;
    const unsigned short* Bh = (const unsigned short*)Bsrc + boff;

    v8f acc[JW];
#pragma unroll
    for (int j = 0; j < JW; ++j) acc[j] = (v8f){0.f,0.f,0.f,0.f,0.f,0.f,0.f,0.f};
    const int ti = wv & 3, jbase = (wv >> 2) * JW;

    auto loadTiles = [&](int buf, int k0) {
        // ---- A tile 64x32 bf16 ----
        if constexpr (CF & CF_GATHER) {
            // gathered rows (token indirection, -1 => zero fill): sync loads
#pragma unroll
            for (int it = 0; it < 4; ++it) {
                int li = tid + it * 256;
                int r = li >> 4, c2 = li & 15;
                int t = asg_tok[blockRow + r];
                unsigned uv = (t >= 0) ? *(const unsigned*)(Ab + (long)t * lda + k0 + c2 * 2) : 0u;
                *(unsigned*)&As[buf][r][c2 * 2] = uv;
            }
        } else {
            // one async 16B DMA per lane: global -> LDS, no VGPR round-trip
            int r = tid >> 2, c = tid & 3;
            long row = blockRow + r;
            unsigned voff = (unsigned)((row * lda + (long)k0 + c * 8) * 2);
            unsigned lofs = (unsigned)(uintptr_t)(&As[buf][r][c * 8]);
            asm volatile("global_load_async_to_lds_b128 %0, %1, %2"
                         :: "v"(lofs), "v"(voff), "s"((unsigned long)Ab)
                         : "memory");
        }
        // ---- B tile 32xNM -> Bs[n][k] (N-major, f32->bf16 convert) ----
        {
            constexpr int PER = NM / 8;
            int k = tid >> 3, n0 = (tid & 7) * PER;
#pragma unroll
            for (int it = 0; it < PER; ++it) {
                int n = n0 + it;
                unsigned short hv;
                if constexpr (CF & CF_B_BF16)
                    hv = Bh[(long)(k0 + k) * ldb + blockCol + n];
                else
                    hv = f2bf(Bf[(long)(k0 + k) * ldb + blockCol + n]);
                Bs[buf][n][k] = hv;
            }
        }
    };

    loadTiles(0, 0);
    if constexpr (!(CF & CF_GATHER)) asm volatile("s_wait_asynccnt 0" ::: "memory");
    __syncthreads();

    int cb = 0;
    for (int k0 = 0; k0 < Kdim; k0 += 32) {
        int nb = cb ^ 1;
        if (k0 + 32 < Kdim) loadTiles(nb, k0 + 32);     // prefetch next tile
        // ---- compute on current buffer ----
        v16bf a;
        {
            int m = ti * 16 + L16;
#pragma unroll
            for (int r = 0; r < 8; ++r) {
                int kb = 2 * r + ((r >= 4) ? 8 : 0) + 8 * hiL;
                put2(a, 2 * r, *(const unsigned*)&As[cb][m][kb]);
            }
        }
#pragma unroll
        for (int j = 0; j < JW; ++j) {
            v16bf b;
            int n = (jbase + j) * 16 + L16;
#pragma unroll
            for (int r = 0; r < 8; ++r) {
                int kk = 2 * r + ((r >= 4) ? 8 : 0) + 8 * hiL;
                put2(b, 2 * r, *(const unsigned*)&Bs[cb][n][kk]);
            }
            acc[j] = __builtin_amdgcn_wmma_f32_16x16x32_bf16(false, a, false, b, (short)0, acc[j], false, false);
        }
        if constexpr (!(CF & CF_GATHER)) asm volatile("s_wait_asynccnt 0" ::: "memory");
        __syncthreads();
        cb = nb;
    }

    float*          Cfp = Cf  ? Cf  + (long)zb * sCb  + (long)zh * sCh  : nullptr;
    unsigned short* Cbp = Cbf ? Cbf + (long)zb * sCbb + (long)zh * sCbh : nullptr;
#pragma unroll
    for (int j = 0; j < JW; ++j) {
        long ng = blockCol + (long)(jbase + j) * 16 + L16;
        float bv = biasp ? biasp[ng] : 0.f;
#pragma unroll
        for (int r = 0; r < 8; ++r) {
            long mg = blockRow + ti * 16 + r + 8 * hiL;
            float val = acc[j][r] + bv;
            if constexpr (CF & CF_GELU)   val = 0.5f * val * (1.f + erff(val * 0.7071067811865475f));
            if constexpr (CF & CF_ELU1)   val = (val > 0.f) ? (val + 1.f) : expf(val);
            if constexpr (CF & CF_ROWMUL) val *= rowptr[rb * zb + rh * zh + mg * rstride];
            if constexpr (CF & CF_ADDRES) val += res[mg * ldres + ng];
            if constexpr (CF & CF_STOREF)  Cfp[mg * ldc + ng] = val;
            if constexpr (CF & CF_STOREBF) Cbp[mg * ldcb + ng] = f2bf(val);
            if constexpr (CF & CF_SCATTER) {
                int t = asg_tok[mg];
                if (t >= 0)
                    ybuf[(long)asg_slot[mg] * yslot_stride + (long)t * ldc + ng] = asg_w[mg] * val;
            }
        }
    }
}

// ---------------------------------------------------------------------------
// LayerNorm over D=1024, one block per token, output bf16
// ---------------------------------------------------------------------------
__global__ void ln_kernel(const float* __restrict__ x, const float* __restrict__ g,
                          const float* __restrict__ b, unsigned short* __restrict__ out)
{
    __shared__ float red[256];
    const long t = blockIdx.x; const int tid = threadIdx.x;
    const float* xr = x + t * 1024;
    float s = 0.f;
    for (int i = tid; i < 1024; i += 256) s += xr[i];
    red[tid] = s; __syncthreads();
    for (int o = 128; o > 0; o >>= 1) { if (tid < o) red[tid] += red[tid + o]; __syncthreads(); }
    float mean = red[0] / 1024.f; __syncthreads();
    float v = 0.f;
    for (int i = tid; i < 1024; i += 256) { float d = xr[i] - mean; v += d * d; }
    red[tid] = v; __syncthreads();
    for (int o = 128; o > 0; o >>= 1) { if (tid < o) red[tid] += red[tid + o]; __syncthreads(); }
    float rstd = rsqrtf(red[0] / 1024.f + 1e-5f);
    unsigned short* orow = out + t * 1024;
    for (int i = tid; i < 1024; i += 256) orow[i] = f2bf((xr[i] - mean) * rstd * g[i] + b[i]);
}

__global__ void maskrow_kernel(const float* __restrict__ m, float* __restrict__ o) {
    int i = blockIdx.x * 256 + threadIdx.x;   // 65536 rows [B,S,H]
    o[i] = m[i >> 4];
}

// phi_k [B,S,H,K] -> phi_kT [B*H, K, S]
__global__ void transpose_phik(const unsigned short* __restrict__ in, unsigned short* __restrict__ out) {
    long o = (long)blockIdx.x * 256 + threadIdx.x;  // 4194304
    int s = o & 2047; int k = (o >> 11) & 63; int bh = o >> 17;
    out[o] = in[((long)(bh >> 4) * 2048 + s) * 1024 + (bh & 15) * 64 + k];
}

// ksum[bh,k] = sum_s phi_kT[bh,k,s]
__global__ void ksum_kernel(const unsigned short* __restrict__ phikT, float* __restrict__ ksum) {
    __shared__ float red[256];
    int bh = blockIdx.x; int k = threadIdx.x & 63; int sc = threadIdx.x >> 6;
    const unsigned short* row = phikT + ((long)bh * 64 + k) * 2048;
    float s = 0.f;
    for (int i = sc * 512; i < sc * 512 + 512; ++i) s += bf2f(row[i]);
    red[threadIdx.x] = s; __syncthreads();
    if (threadIdx.x < 64) ksum[bh * 64 + k] = red[k] + red[64 + k] + red[128 + k] + red[192 + k];
}

// zinv[b,s,h] = 1/(phi_q[b,s,h,:].ksum[b,h,:] + 1e-6)
__global__ void zinv_kernel(const unsigned short* __restrict__ phiq, const float* __restrict__ ksum,
                            float* __restrict__ zinv) {
    long r = (long)blockIdx.x * 256 + threadIdx.x;  // 65536
    int h = r & 15; long bs = r >> 4; int b = (int)(bs >> 11);
    const unsigned short* qr = phiq + r * 64;
    const float* ks = ksum + ((long)b * 16 + h) * 64;
    float z = 0.f;
    for (int k = 0; k < 64; ++k) z += bf2f(qr[k]) * ks[k];
    zinv[r] = 1.f / (z + 1e-6f);
}

// gate: logits -> softmax -> top2 -> normalized weights + counts
__global__ void gate_kernel(const unsigned short* __restrict__ xn2, const float* __restrict__ gw,
                            const float* __restrict__ gb, int* __restrict__ tok_e,
                            float* __restrict__ tok_w, int* __restrict__ counts) {
    __shared__ float red[256 * 8];
    int t = blockIdx.x, tid = threadIdx.x;
    float acc[8] = {0,0,0,0,0,0,0,0};
    const unsigned short* xr = xn2 + (long)t * 1024;
    for (int d = tid; d < 1024; d += 256) {
        float xv = bf2f(xr[d]);
        const float* gr = gw + (long)d * 8;
        for (int e = 0; e < 8; ++e) acc[e] += xv * gr[e];
    }
    for (int e = 0; e < 8; ++e) red[tid * 8 + e] = acc[e];
    __syncthreads();
    for (int o = 128; o > 0; o >>= 1) {
        if (tid < o) for (int e = 0; e < 8; ++e) red[tid * 8 + e] += red[(tid + o) * 8 + e];
        __syncthreads();
    }
    if (tid == 0) {
        float lg[8]; float mx = -1e30f;
        for (int e = 0; e < 8; ++e) { lg[e] = red[e] + gb[e]; mx = fmaxf(mx, lg[e]); }
        float s = 0.f;
        for (int e = 0; e < 8; ++e) { lg[e] = expf(lg[e] - mx); s += lg[e]; }
        for (int e = 0; e < 8; ++e) lg[e] /= s;
        int e0 = 0;
        for (int e = 1; e < 8; ++e) if (lg[e] > lg[e0]) e0 = e;
        int e1 = -1;
        for (int e = 0; e < 8; ++e) { if (e == e0) continue; if (e1 < 0 || lg[e] > lg[e1]) e1 = e; }
        float w0 = lg[e0], w1 = lg[e1], den = w0 + w1 + 1e-9f;
        tok_e[t * 2] = e0; tok_e[t * 2 + 1] = e1;
        tok_w[t * 2] = w0 / den; tok_w[t * 2 + 1] = w1 / den;
        atomicAdd(&counts[e0], 1); atomicAdd(&counts[e1], 1);
    }
}

__global__ void offsets_kernel(const int* __restrict__ counts, int* __restrict__ offv,
                               int* __restrict__ cursor) {
    if (threadIdx.x == 0) {
        int o = 0;
        for (int e = 0; e < 8; ++e) { offv[e] = o; cursor[e] = o; o += (counts[e] + 63) & ~63; }
        offv[8] = o;
    }
}

__global__ void assign_kernel(const int* __restrict__ tok_e, const float* __restrict__ tok_w,
                              int* __restrict__ cursor, int* __restrict__ asg_tok,
                              float* __restrict__ asg_w, int* __restrict__ asg_slot) {
    int t = blockIdx.x * 256 + threadIdx.x;
    if (t >= 4096) return;
    for (int s = 0; s < 2; ++s) {
        int e = tok_e[t * 2 + s];
        int pos = atomicAdd(&cursor[e], 1);
        asg_tok[pos] = t; asg_w[pos] = tok_w[t * 2 + s]; asg_slot[pos] = s;
    }
}

__global__ void init_kernel(int* __restrict__ counts, int* __restrict__ asg_tok) {
    int i = blockIdx.x * 256 + threadIdx.x;
    if (i < 8) counts[i] = 0;
    if (i < 8704) asg_tok[i] = -1;
}

__global__ void final_kernel(const float* __restrict__ x1, const float* __restrict__ ybuf,
                             float* __restrict__ out) {
    long i = (long)blockIdx.x * 256 + threadIdx.x;  // 4194304
    out[i] = x1[i] + ybuf[i] + ybuf[4194304 + i];
}

// ---------------------------------------------------------------------------
extern "C" void kernel_launch(void* const* d_in, const int* in_sizes, int n_in,
                              void* d_out, int out_size, void* d_ws, size_t ws_size,
                              hipStream_t stream) {
    const float* x    = (const float*)d_in[0];
    const float* am   = (const float*)d_in[1];
    const float* ln1g = (const float*)d_in[2];
    const float* ln1b = (const float*)d_in[3];
    const float* wq   = (const float*)d_in[4];
    const float* bq   = (const float*)d_in[5];
    const float* wk   = (const float*)d_in[6];
    const float* bk   = (const float*)d_in[7];
    const float* wv   = (const float*)d_in[8];
    const float* bv   = (const float*)d_in[9];
    const float* wo   = (const float*)d_in[10];
    const float* bo   = (const float*)d_in[11];
    const float* pq   = (const float*)d_in[12];
    const float* pk   = (const float*)d_in[13];
    const float* ln2g = (const float*)d_in[14];
    const float* ln2b = (const float*)d_in[15];
    const float* gw   = (const float*)d_in[16];
    const float* gb   = (const float*)d_in[17];
    const float* ew1  = (const float*)d_in[18];
    const float* eb1  = (const float*)d_in[19];
    const float* ew2  = (const float*)d_in[20];
    const float* eb2  = (const float*)d_in[21];

    const size_t TD = 4194304;  // T*D = 4096*1024
    char* base = (char*)d_ws; size_t cur = 0;
    auto alloc = [&](size_t bytes) -> void* {
        void* p = base + cur; cur = (cur + bytes + 255) & ~(size_t)255; return p;
    };
    unsigned short* xn1    = (unsigned short*)alloc(TD * 2);
    unsigned short* qb     = (unsigned short*)alloc(TD * 2);
    unsigned short* kb     = (unsigned short*)alloc(TD * 2);
    unsigned short* vb     = (unsigned short*)alloc(TD * 2);
    unsigned short* phiq   = (unsigned short*)alloc(TD * 2);
    unsigned short* phik   = (unsigned short*)alloc(TD * 2);
    unsigned short* phikT  = (unsigned short*)alloc(TD * 2);
    unsigned short* kvb    = (unsigned short*)alloc((size_t)32 * 4096 * 2);
    unsigned short* attnb  = (unsigned short*)alloc(TD * 2);
    float*          x1     = (float*)alloc(TD * 4);
    unsigned short* xn2    = (unsigned short*)alloc(TD * 2);
    float*          ksum   = (float*)alloc(2048 * 4);
    float*          zinv   = (float*)alloc(65536 * 4);
    float*          mrow   = (float*)alloc(65536 * 4);
    int*            tok_e  = (int*)alloc(8192 * 4);
    float*          tok_w  = (float*)alloc(8192 * 4);
    int*            counts = (int*)alloc(8 * 4);
    int*            offv   = (int*)alloc(9 * 4);
    int*            cursor = (int*)alloc(8 * 4);
    int*            asg_t  = (int*)alloc(8704 * 4);
    float*          asg_w  = (float*)alloc(8704 * 4);
    int*            asg_s  = (int*)alloc(8704 * 4);
    unsigned short* Hbuf   = (unsigned short*)alloc((size_t)8704 * 4096 * 2);
    float*          ybuf   = (float*)alloc((size_t)2 * TD * 4);

    init_kernel<<<34, 256, 0, stream>>>(counts, asg_t);
    ln_kernel<<<4096, 256, 0, stream>>>(x, ln1g, ln1b, xn1);

    // Q/K/V: [4096,1024] = xn1 @ W + b  -> bf16   (macro 64x128)
    gemm_k<CF_STOREBF,4><<<dim3(64,8,1),256,0,stream>>>(
        xn1,1024,0,0, wq,1024,0,0, bq, nullptr,0,0,0, qb,1024,0,0,
        nullptr,0,0,0, nullptr,0, 1024,1, nullptr,nullptr,nullptr,nullptr, nullptr,0, 0,0);
    gemm_k<CF_STOREBF,4><<<dim3(64,8,1),256,0,stream>>>(
        xn1,1024,0,0, wk,1024,0,0, bk, nullptr,0,0,0, kb,1024,0,0,
        nullptr,0,0,0, nullptr,0, 1024,1, nullptr,nullptr,nullptr,nullptr, nullptr,0, 0,0);
    gemm_k<CF_STOREBF,4><<<dim3(64,8,1),256,0,stream>>>(
        xn1,1024,0,0, wv,1024,0,0, bv, nullptr,0,0,0, vb,1024,0,0,
        nullptr,0,0,0, nullptr,0, 1024,1, nullptr,nullptr,nullptr,nullptr, nullptr,0, 0,0);

    maskrow_kernel<<<256, 256, 0, stream>>>(am, mrow);

    // phi_q = elu(q@pq)+1 : M=65536, N=64, K=64  (macro 64x64)
    gemm_k<CF_ELU1|CF_STOREBF,2><<<dim3(1024,1,1),256,0,stream>>>(
        qb,64,0,0, pq,64,0,0, nullptr, nullptr,0,0,0, phiq,64,0,0,
        nullptr,0,0,0, nullptr,0, 64,1, nullptr,nullptr,nullptr,nullptr, nullptr,0, 0,0);
    // phi_k = (elu(k@pk)+1) * mask
    gemm_k<CF_ELU1|CF_ROWMUL|CF_STOREBF,2><<<dim3(1024,1,1),256,0,stream>>>(
        kb,64,0,0, pk,64,0,0, nullptr, nullptr,0,0,0, phik,64,0,0,
        mrow,1,0,0, nullptr,0, 64,1, nullptr,nullptr,nullptr,nullptr, nullptr,0, 0,0);

    transpose_phik<<<16384, 256, 0, stream>>>(phik, phikT);
    ksum_kernel<<<32, 256, 0, stream>>>(phikT, ksum);

    // kv[b,h] = phi_k^T @ v : M=64, N=64, K=2048, batch 32 (Hdiv=16)
    gemm_k<CF_B_BF16|CF_STOREBF,2><<<dim3(1,1,32),256,0,stream>>>(
        phikT,2048,2097152,131072, vb,1024,2097152,64, nullptr,
        nullptr,0,0,0, kvb,64,65536,4096,
        nullptr,0,0,0, nullptr,0, 2048,16, nullptr,nullptr,nullptr,nullptr, nullptr,0, 0,0);

    zinv_kernel<<<256, 256, 0, stream>>>(phiq, ksum, zinv);

    // attn[b,:,h,:] = (phi_q @ kv) * zinv : M=2048, N=64, K=64, batch 32
    gemm_k<CF_B_BF16|CF_ROWMUL|CF_STOREBF,2><<<dim3(32,1,32),256,0,stream>>>(
        phiq,1024,2097152,64, kvb,64,65536,4096, nullptr,
        nullptr,0,0,0, attnb,1024,2097152,64,
        zinv,16,32768,1, nullptr,0, 64,16, nullptr,nullptr,nullptr,nullptr, nullptr,0, 0,0);

    // x1 = x + attn @ wo + bo   (macro 64x128)
    gemm_k<CF_ADDRES|CF_STOREF,4><<<dim3(64,8,1),256,0,stream>>>(
        attnb,1024,0,0, wo,1024,0,0, bo, x1,1024,0,0, nullptr,0,0,0,
        nullptr,0,0,0, x,1024, 1024,1, nullptr,nullptr,nullptr,nullptr, nullptr,0, 0,0);

    ln_kernel<<<4096, 256, 0, stream>>>(x1, ln2g, ln2b, xn2);
    gate_kernel<<<4096, 256, 0, stream>>>(xn2, gw, gb, tok_e, tok_w, counts);
    offsets_kernel<<<1, 32, 0, stream>>>(counts, offv, cursor);
    assign_kernel<<<16, 256, 0, stream>>>(tok_e, tok_w, cursor, asg_t, asg_w, asg_s);

    // Expert GEMM1: H = gelu(gather(xn2) @ w1[e] + b1[e]) -> bf16 [8704,4096]
    gemm_k<CF_GATHER|CF_EXPERT|CF_GELU|CF_STOREBF,4><<<dim3(136,32,1),256,0,stream>>>(
        xn2,1024,0,0, ew1,4096,0,0, eb1, nullptr,0,0,0, Hbuf,4096,0,0,
        nullptr,0,0,0, nullptr,0, 1024,1, offv,asg_t,asg_w,asg_s, nullptr,0,
        (long)1024*4096, 4096);
    // Expert GEMM2: scatter w * (H @ w2[e] + b2[e]) into per-slot buffers
    gemm_k<CF_EXPERT|CF_SCATTER,4><<<dim3(136,8,1),256,0,stream>>>(
        Hbuf,4096,0,0, ew2,1024,0,0, eb2, nullptr,1024,0,0, nullptr,0,0,0,
        nullptr,0,0,0, nullptr,0, 4096,1, offv,asg_t,asg_w,asg_s, ybuf,(long)TD,
        (long)4096*1024, 1024);

    final_kernel<<<16384, 256, 0, stream>>>(x1, ybuf, (float*)d_out);
}
